// GATLayer_3435973837172
// MI455X (gfx1250) — compile-verified, matrix operand order checked by reading, not run on hardware
//
#include <hip/hip_runtime.h>

#define IN_C   256
#define OUT_C  64
#define NEG_SLOPE 0.2f

typedef __attribute__((ext_vector_type(16))) __bf16 v16bf;
typedef __attribute__((ext_vector_type(8)))  float  v8f;

union BF16x16 {
    v16bf  v;
    __bf16 b[16];
};

// ---------------------------------------------------------------------------
// Pack W (OUT_C x IN_C fp32, row-major) into per-lane WMMA B-fragment layout,
// bf16 (native HW convert, RNE). Layout: [tile t(4)][kstep s(8)][lane(32)] ->
// 16 bf16 (32 bytes), where lane n = t*16 + (lane&15), half = lane>>4, and
// VGPR r holds K = s*32 + half*8 + (r&3)*2 + (r>>2)*16  (ISA 7.12.2).
// ---------------------------------------------------------------------------
__global__ void pack_w_kernel(const float* __restrict__ W,
                              __bf16* __restrict__ Wp) {
    int item = blockIdx.x * blockDim.x + threadIdx.x;   // 0..1023
    if (item >= 4 * 8 * 32) return;
    int lane = item & 31;
    int s    = (item >> 5) & 7;
    int t    = item >> 8;
    int n    = t * 16 + (lane & 15);
    int half = lane >> 4;
    __bf16* dst = Wp + (size_t)item * 16;
    #pragma unroll
    for (int r = 0; r < 8; ++r) {
        int k = s * 32 + half * 8 + (r & 3) * 2 + ((r >> 2) << 4);
        dst[2 * r]     = (__bf16)W[n * IN_C + k];
        dst[2 * r + 1] = (__bf16)W[n * IN_C + k + 1];
    }
}

__device__ __forceinline__ void load_A_frag(const float* __restrict__ xrow,
                                            int s, int half, BF16x16& A) {
    int kbase = s * 32 + half * 8;
    #pragma unroll
    for (int r = 0; r < 8; ++r) {
        int k = kbase + (r & 3) * 2 + ((r >> 2) << 4);
        float2 xf = *(const float2*)(xrow + k);
        A.b[2 * r]     = (__bf16)xf.x;   // v_cvt bf16 (RNE in HW)
        A.b[2 * r + 1] = (__bf16)xf.y;
    }
}

// ---------------------------------------------------------------------------
// h = x @ W^T via v_wmma_f32_16x16x32_bf16, software-pipelined: fragments for
// step s+1 are fetched before the 4 WMMAs of step s issue, so global loads
// overlap matrix-core execution instead of s_wait-stalling in front of it.
// Block = 256 threads = 8 waves; wave w computes rows [blk*128 + w*16, +16)
// across all 64 output columns (4 accumulator tiles). K = 256 -> 8 steps.
// ---------------------------------------------------------------------------
__global__ __launch_bounds__(256) void gemm_h_kernel(
        const float* __restrict__ x,
        const __bf16* __restrict__ Wp,
        float* __restrict__ h, int nNodes) {
    int wave = threadIdx.x >> 5;
    int lane = threadIdx.x & 31;
    int half = lane >> 4;
    int mloc = lane & 15;
    int row0 = blockIdx.x * 128 + wave * 16;

    int rowA = row0 + mloc;
    if (rowA >= nNodes) rowA = nNodes - 1;   // clamp reads: EXEC stays all-1s
    const float* xrow = x + (size_t)rowA * IN_C;

    // Warm this row's cachelines (global_prefetch_b8), 1 KB per row.
    __builtin_prefetch(xrow, 0, 3);
    __builtin_prefetch(xrow + 128, 0, 3);

    const v16bf* bp = (const v16bf*)Wp;      // 32B-aligned fragments

    v8f acc0 = {}, acc1 = {}, acc2 = {}, acc3 = {};

    // Prologue: fragments for step 0.
    BF16x16 A;
    load_A_frag(xrow, 0, half, A);
    v16bf B0 = bp[(0 * 8 + 0) * 32 + lane];
    v16bf B1 = bp[(1 * 8 + 0) * 32 + lane];
    v16bf B2 = bp[(2 * 8 + 0) * 32 + lane];
    v16bf B3 = bp[(3 * 8 + 0) * 32 + lane];

    #pragma unroll
    for (int s = 0; s < 8; ++s) {
        BF16x16 An;
        v16bf B0n, B1n, B2n, B3n;
        if (s < 7) {
            load_A_frag(xrow, s + 1, half, An);
            B0n = bp[(0 * 8 + s + 1) * 32 + lane];
            B1n = bp[(1 * 8 + s + 1) * 32 + lane];
            B2n = bp[(2 * 8 + s + 1) * 32 + lane];
            B3n = bp[(3 * 8 + s + 1) * 32 + lane];
        }
        acc0 = __builtin_amdgcn_wmma_f32_16x16x32_bf16(false, A.v, false, B0,
                                                       (short)0, acc0, false, false);
        acc1 = __builtin_amdgcn_wmma_f32_16x16x32_bf16(false, A.v, false, B1,
                                                       (short)0, acc1, false, false);
        acc2 = __builtin_amdgcn_wmma_f32_16x16x32_bf16(false, A.v, false, B2,
                                                       (short)0, acc2, false, false);
        acc3 = __builtin_amdgcn_wmma_f32_16x16x32_bf16(false, A.v, false, B3,
                                                       (short)0, acc3, false, false);
        if (s < 7) {
            A = An;
            B0 = B0n; B1 = B1n; B2 = B2n; B3 = B3n;
        }
    }

    // C/D layout: lane -> N = lane&15, VGPR r -> M = (lane>>4)*8 + r
    #pragma unroll
    for (int r = 0; r < 8; ++r) {
        int row = row0 + half * 8 + r;
        if (row < nNodes) {
            float* o = h + (size_t)row * OUT_C + mloc;
            o[0]  = acc0[r];
            o[16] = acc1[r];
            o[32] = acc2[r];
            o[48] = acc3[r];
        }
    }
}

// ---------------------------------------------------------------------------
// Per-node attention scores: s_i = h . a[:64], s_j = h . a[64:].
// One wave per node; lane handles 2 channels; wave32 shfl_xor reduction.
// ---------------------------------------------------------------------------
__global__ __launch_bounds__(256) void score_kernel(
        const float* __restrict__ h, const float* __restrict__ a,
        float* __restrict__ s_i, float* __restrict__ s_j, int n) {
    int wave = threadIdx.x >> 5;
    int lane = threadIdx.x & 31;
    int node = blockIdx.x * 8 + wave;
    if (node >= n) return;
    float2 hv = *(const float2*)(h + (size_t)node * OUT_C + lane * 2);
    float2 ai = *(const float2*)(a + lane * 2);
    float2 aj = *(const float2*)(a + OUT_C + lane * 2);
    float pi = hv.x * ai.x + hv.y * ai.y;
    float pj = hv.x * aj.x + hv.y * aj.y;
    #pragma unroll
    for (int off = 16; off > 0; off >>= 1) {
        pi += __shfl_xor(pi, off, 32);
        pj += __shfl_xor(pj, off, 32);
    }
    if (lane == 0) { s_i[node] = pi; s_j[node] = pj; }
}

__global__ void init_kernel(float* __restrict__ out, float* __restrict__ m,
                            float* __restrict__ denom, int n) {
    int i = blockIdx.x * blockDim.x + threadIdx.x;
    if (i < n * OUT_C) out[i] = 0.0f;
    if (i < n) { m[i] = -__builtin_inff(); denom[i] = 0.0f; }
}

__device__ __forceinline__ void edge_rc(const int* __restrict__ ei, int nE,
                                        int i, int& r, int& c) {
    if (i < nE) { r = ei[i]; c = ei[nE + i]; }
    else        { r = i - nE; c = i - nE; }          // self loops
}

__device__ __forceinline__ float lrelu(float e) {
    return e > 0.0f ? e : NEG_SLOPE * e;
}

__global__ void edge_max_kernel(const int* __restrict__ ei, int nE, int nTot,
                                const float* __restrict__ s_i,
                                const float* __restrict__ s_j,
                                float* __restrict__ m) {
    int i = blockIdx.x * blockDim.x + threadIdx.x;
    if (i >= nTot) return;
    int r, c; edge_rc(ei, nE, i, r, c);
    float e = lrelu(s_i[c] + s_j[r]);
    __hip_atomic_fetch_max(&m[c], e, __ATOMIC_RELAXED, __HIP_MEMORY_SCOPE_AGENT);
}

__global__ void edge_sum_kernel(const int* __restrict__ ei, int nE, int nTot,
                                const float* __restrict__ s_i,
                                const float* __restrict__ s_j,
                                const float* __restrict__ m,
                                float* __restrict__ denom) {
    int i = blockIdx.x * blockDim.x + threadIdx.x;
    if (i >= nTot) return;
    int r, c; edge_rc(ei, nE, i, r, c);
    float e  = lrelu(s_i[c] + s_j[r]);
    float ex = __expf(e - m[c]);
    __hip_atomic_fetch_add(&denom[c], ex, __ATOMIC_RELAXED, __HIP_MEMORY_SCOPE_AGENT);
}

// One wave per edge: coalesced 256B read of h[row], 64 L2-resident float
// atomic adds into out[col].
__global__ __launch_bounds__(256) void edge_aggr_kernel(
        const int* __restrict__ ei, int nE, int nTot,
        const float* __restrict__ s_i, const float* __restrict__ s_j,
        const float* __restrict__ m, const float* __restrict__ denom,
        const float* __restrict__ h, float* __restrict__ out) {
    int wave = threadIdx.x >> 5;
    int lane = threadIdx.x & 31;
    int i = blockIdx.x * 8 + wave;
    if (i >= nTot) return;
    int r, c; edge_rc(ei, nE, i, r, c);
    float e = lrelu(s_i[c] + s_j[r]);
    float alpha = __expf(e - m[c]) / denom[c];
    float2 hv = *(const float2*)(h + (size_t)r * OUT_C + lane * 2);
    float* o = out + (size_t)c * OUT_C + lane * 2;
    __hip_atomic_fetch_add(&o[0], alpha * hv.x, __ATOMIC_RELAXED, __HIP_MEMORY_SCOPE_AGENT);
    __hip_atomic_fetch_add(&o[1], alpha * hv.y, __ATOMIC_RELAXED, __HIP_MEMORY_SCOPE_AGENT);
}

extern "C" void kernel_launch(void* const* d_in, const int* in_sizes, int n_in,
                              void* d_out, int out_size, void* d_ws, size_t ws_size,
                              hipStream_t stream) {
    const float* x  = (const float*)d_in[0];
    const int*   ei = (const int*)d_in[1];
    const float* W  = (const float*)d_in[2];
    const float* a  = (const float*)d_in[3];

    const int N  = in_sizes[0] / IN_C;     // 100000
    const int nE = in_sizes[1] / 2;        // 3200000
    const int nTot = nE + N;               // + self loops

    float* out = (float*)d_out;

    // Workspace layout (floats): h[N*64] | s_i[N] | s_j[N] | m[N] | denom[N] | Wp
    float* ws    = (float*)d_ws;
    float* h     = ws;
    float* s_i   = h + (size_t)N * OUT_C;
    float* s_j   = s_i + N;
    float* mbuf  = s_j + N;
    float* denom = mbuf + N;
    __bf16* Wp   = (__bf16*)(denom + N);   // 32 KB, 32B-aligned

    init_kernel<<<(N * OUT_C + 255) / 256, 256, 0, stream>>>(out, mbuf, denom, N);
    pack_w_kernel<<<4, 256, 0, stream>>>(W, Wp);
    gemm_h_kernel<<<(N + 127) / 128, 256, 0, stream>>>(x, Wp, h, N);
    score_kernel<<<(N + 7) / 8, 256, 0, stream>>>(h, a, s_i, s_j, N);
    edge_max_kernel<<<(nTot + 255) / 256, 256, 0, stream>>>(ei, nE, nTot, s_i, s_j, mbuf);
    edge_sum_kernel<<<(nTot + 255) / 256, 256, 0, stream>>>(ei, nE, nTot, s_i, s_j, mbuf, denom);
    edge_aggr_kernel<<<(nTot + 7) / 8, 256, 0, stream>>>(ei, nE, nTot, s_i, s_j, mbuf, denom, h, out);
}
